// CausalMultiHeadAttention_88794153878163
// MI455X (gfx1250) — compile-verified, matrix operand order checked by reading, not run on hardware
//
#include <hip/hip_runtime.h>
#include <stdint.h>

#define BATCH  2
#define SEQ    2048
#define DMODEL 1024
#define NHEAD  16
#define DHEAD  64
#define MROWS  (BATCH * SEQ)   // 4096

typedef __attribute__((ext_vector_type(16))) __bf16 v16bf;
typedef __attribute__((ext_vector_type(8)))  __bf16 v8bf;
typedef __attribute__((ext_vector_type(8)))  float  v8f;

typedef int v4i __attribute__((vector_size(4 * sizeof(int))));
typedef __attribute__((address_space(1))) v4i gas_v4i;
typedef __attribute__((address_space(3))) v4i las_v4i;

#define HAVE_ASYNC_LDS __has_builtin(__builtin_amdgcn_global_load_async_to_lds_b128)
#if !HAVE_ASYNC_LDS
#pragma message("CDNA5: global_load_async_to_lds builtin NOT available -> sync LDS staging fallback")
#else
#pragma message("CDNA5: using global_load_async_to_lds_b128 + s_wait_asynccnt path")
#endif

__device__ __forceinline__ __bf16 f2bf(float f) {
  union { float f; uint32_t u; } v; v.f = f;
  uint32_t r = v.u + 0x7FFFu + ((v.u >> 16) & 1u);   // round-to-nearest-even
  uint16_t h = (uint16_t)(r >> 16);
  __bf16 o;
  __builtin_memcpy(&o, &h, sizeof(o));
  return o;
}

// 16-byte global -> LDS copy, async (ASYNCcnt) when available.
__device__ __forceinline__ void cp16_g2l(const __bf16* g, __bf16* l) {
#if HAVE_ASYNC_LDS
  __builtin_amdgcn_global_load_async_to_lds_b128(
      (gas_v4i*)g, (las_v4i*)l, 0, 0);
#else
  *(v8bf*)l = *(const v8bf*)g;
#endif
}

__device__ __forceinline__ void async_wait_all() {
#if HAVE_ASYNC_LDS
#if __has_builtin(__builtin_amdgcn_s_wait_asynccnt)
  __builtin_amdgcn_s_wait_asynccnt(0);
#else
  asm volatile("s_wait_asynccnt 0x0" ::: "memory");
#endif
#endif
}

// ---------------------------------------------------------------- convert ---
__global__ __launch_bounds__(256) void to_bf16_kernel(
    const float* __restrict__ src, __bf16* __restrict__ dst, int n) {
  int i = blockIdx.x * blockDim.x + threadIdx.x;
  int stride = gridDim.x * blockDim.x;
  for (; i < n; i += stride) dst[i] = f2bf(src[i]);
}

// ------------------------------------------------------------------- GEMM ---
// C[M=4096, N=1024] = A[M,K=1024] * B[N,K]^T   (both operands K-contiguous)
// Block: 256 threads = 8 waves (4x2), block tile 128x128, wave tile 32x64.
// Double-buffered LDS slabs filled with async global->LDS copies.
// mode 0: store Q bf16 [B,H,S,DK], scaled by 1/sqrt(DK)
// mode 1: store K bf16 [B,H,S,DK]
// mode 2: store V bf16 transposed [B,H,DK,S]
// mode 3: store fp32 row-major [M,N] (final projection -> d_out)
__device__ __forceinline__ void stage_slabs(
    const __bf16* __restrict__ A, const __bf16* __restrict__ Bm,
    __bf16* lA, __bf16* lB, int m0, int n0, int k0, int tid) {
  // 128 rows x 32 cols bf16 per slab = 8KB; 256 threads x 2 chunks x 16B
  int c = tid;
  int row = c >> 2, q = (c & 3) * 8;
  cp16_g2l(&A[(size_t)(m0 + row) * DMODEL + k0 + q], &lA[row * 32 + q]);
  cp16_g2l(&Bm[(size_t)(n0 + row) * DMODEL + k0 + q], &lB[row * 32 + q]);
  c = tid + 256;
  row = c >> 2; q = (c & 3) * 8;
  cp16_g2l(&A[(size_t)(m0 + row) * DMODEL + k0 + q], &lA[row * 32 + q]);
  cp16_g2l(&Bm[(size_t)(n0 + row) * DMODEL + k0 + q], &lB[row * 32 + q]);
}

__global__ __launch_bounds__(256) void gemm_nt_kernel(
    const __bf16* __restrict__ A, const __bf16* __restrict__ Bm,
    __bf16* __restrict__ outb, float* __restrict__ outf, int mode) {
  __shared__ __bf16 lA[2][128 * 32];
  __shared__ __bf16 lB[2][128 * 32];

  const int tid  = threadIdx.x;
  const int wave = tid >> 5;
  const int lane = tid & 31;
  const int ln   = lane & 15;
  const int kh   = (lane < 16) ? 0 : 8;       // K-offset of this lane half
  const int m0   = blockIdx.x * 128;
  const int n0   = blockIdx.y * 128;
  const int wm   = wave >> 1;                 // 0..3  (M)
  const int wn   = wave & 1;                  // 0..1  (N)

  v8f acc[2][4];
#pragma unroll
  for (int mi = 0; mi < 2; ++mi)
#pragma unroll
    for (int ni = 0; ni < 4; ++ni) acc[mi][ni] = {};

  stage_slabs(A, Bm, lA[0], lB[0], m0, n0, 0, tid);

  for (int k0 = 0; k0 < DMODEL; k0 += 32) {
    const int buf = (k0 >> 5) & 1;
    async_wait_all();          // this slab's async copies landed in LDS
    __syncthreads();           // ... and are visible to all waves
    if (k0 + 32 < DMODEL)      // overlap next slab's DMA with the WMMAs below
      stage_slabs(A, Bm, lA[buf ^ 1], lB[buf ^ 1], m0, n0, k0 + 32, tid);

    v16bf aF[2], bF[4];
#pragma unroll
    for (int mi = 0; mi < 2; ++mi) {
      const __bf16* p = &lA[buf][(wm * 32 + mi * 16 + ln) * 32 + kh];
      v8bf lo = *(const v8bf*)p;
      v8bf hi = *(const v8bf*)(p + 16);
#pragma unroll
      for (int e = 0; e < 8; ++e) { aF[mi][e] = lo[e]; aF[mi][e + 8] = hi[e]; }
    }
#pragma unroll
    for (int ni = 0; ni < 4; ++ni) {
      const __bf16* p = &lB[buf][(wn * 64 + ni * 16 + ln) * 32 + kh];
      v8bf lo = *(const v8bf*)p;
      v8bf hi = *(const v8bf*)(p + 16);
#pragma unroll
      for (int e = 0; e < 8; ++e) { bF[ni][e] = lo[e]; bF[ni][e + 8] = hi[e]; }
    }
#pragma unroll
    for (int mi = 0; mi < 2; ++mi)
#pragma unroll
      for (int ni = 0; ni < 4; ++ni)
        acc[mi][ni] = __builtin_amdgcn_wmma_f32_16x16x32_bf16(
            false, aF[mi], false, bF[ni], (short)0, acc[mi][ni], false, false);
  }

  // epilogue: C layout -> lane holds column n = ln, rows r + kh
#pragma unroll
  for (int mi = 0; mi < 2; ++mi)
#pragma unroll
    for (int ni = 0; ni < 4; ++ni)
#pragma unroll
      for (int r = 0; r < 8; ++r) {
        const int row = m0 + wm * 32 + mi * 16 + r + kh;   // 0..4095
        const int col = n0 + wn * 64 + ni * 16 + ln;       // 0..1023
        const float v = acc[mi][ni][r];
        const int b = row >> 11, s = row & 2047;
        const int h = col >> 6,  d = col & 63;
        if (mode == 0) {
          outb[((size_t)(b * NHEAD + h) * SEQ + s) * DHEAD + d] = f2bf(v * 0.125f);
        } else if (mode == 1) {
          outb[((size_t)(b * NHEAD + h) * SEQ + s) * DHEAD + d] = f2bf(v);
        } else if (mode == 2) {
          outb[((size_t)(b * NHEAD + h) * DHEAD + d) * SEQ + s] = f2bf(v);
        } else {
          outf[(size_t)row * DMODEL + col] = v;
        }
      }
}

// -------------------------------------------------------- flash attention ---
// One wave per 16-query tile. Computes S^T = K*Q^T so softmax reductions are
// in-register + one shfl_xor(16), and P^T in C-layout IS the B-fragment
// layout for O^T = V^T * P^T (no cross-lane data movement).
__global__ __launch_bounds__(256) void attn_kernel(
    const __bf16* __restrict__ Q,   // [B*H, S, DK], pre-scaled by 1/8
    const __bf16* __restrict__ K,   // [B*H, S, DK]
    const __bf16* __restrict__ Vt,  // [B*H, DK, S]
    __bf16* __restrict__ O) {       // [B*S, DMODEL]
  const int tid  = threadIdx.x;
  const int wave = tid >> 5;
  const int lane = tid & 31;
  const int ln   = lane & 15;
  const int kh   = (lane < 16) ? 0 : 8;
  const int tile = blockIdx.x * 8 + wave;   // 4096 tiles total
  const int bh   = tile >> 7;               // 0..31
  const int qt   = tile & 127;
  const int q0   = qt * 16;

  const __bf16* Qh = Q  + (size_t)bh * SEQ * DHEAD;
  const __bf16* Kh = K  + (size_t)bh * SEQ * DHEAD;
  const __bf16* Vh = Vt + (size_t)bh * DHEAD * SEQ;

  // Q fragments (B operand), lane -> query column q0+ln, two 32-wide d steps
  v16bf bq[2];
  const int qrow = q0 + ln;
#pragma unroll
  for (int ds = 0; ds < 2; ++ds) {
    const __bf16* p = Qh + (size_t)qrow * DHEAD + ds * 32 + kh;
    v8bf lo = *(const v8bf*)p;
    v8bf hi = *(const v8bf*)(p + 16);
#pragma unroll
    for (int e = 0; e < 8; ++e) { bq[ds][e] = lo[e]; bq[ds][e + 8] = hi[e]; }
  }

  v8f accO[4];
#pragma unroll
  for (int t = 0; t < 4; ++t) accO[t] = {};
  float m_run = -1e30f, l_run = 0.0f;
  const int myq = q0 + ln;                // this lane's query index

  const int nblocks = (q0 >> 5) + 1;      // key blocks of 32, causal
  for (int kb = 0; kb < nblocks; ++kb) {
    const int k0 = kb * 32;

    // scores^T: two 16-key subtiles, K rows as A operand
    v8f sc[2];
    sc[0] = {}; sc[1] = {};
#pragma unroll
    for (int sub = 0; sub < 2; ++sub)
#pragma unroll
      for (int ds = 0; ds < 2; ++ds) {
        const __bf16* p = Kh + (size_t)(k0 + sub * 16 + ln) * DHEAD + ds * 32 + kh;
        v8bf lo = *(const v8bf*)p;
        v8bf hi = *(const v8bf*)(p + 16);
        v16bf aK;
#pragma unroll
        for (int e = 0; e < 8; ++e) { aK[e] = lo[e]; aK[e + 8] = hi[e]; }
        sc[sub] = __builtin_amdgcn_wmma_f32_16x16x32_bf16(
            false, aK, false, bq[ds], (short)0, sc[sub], false, false);
      }

    // causal mask (wave-uniform branch; per-lane selects only)
    if (k0 + 31 > q0) {
#pragma unroll
      for (int sub = 0; sub < 2; ++sub)
#pragma unroll
        for (int r = 0; r < 8; ++r) {
          const int key = k0 + sub * 16 + r + kh;
          if (key > myq) sc[sub][r] = -1e30f;
        }
    }

    // online softmax: row = query = lane column (pair lane^16 holds other keys)
    float mx = -1e30f;
#pragma unroll
    for (int sub = 0; sub < 2; ++sub)
#pragma unroll
      for (int r = 0; r < 8; ++r) mx = fmaxf(mx, sc[sub][r]);
    mx = fmaxf(mx, __shfl_xor(mx, 16));
    const float m_new = fmaxf(m_run, mx);
    const float alpha = __expf(m_run - m_new);

    float sum = 0.0f;
    v16bf bp;                              // P^T already in B-fragment layout
#pragma unroll
    for (int r = 0; r < 8; ++r) {
      const float p0 = __expf(sc[0][r] - m_new);
      const float p1 = __expf(sc[1][r] - m_new);
      sum += p0 + p1;
      bp[r] = f2bf(p0);
      bp[r + 8] = f2bf(p1);
    }
    sum += __shfl_xor(sum, 16);
    l_run = l_run * alpha + sum;
    m_run = m_new;

#pragma unroll
    for (int t = 0; t < 4; ++t)
#pragma unroll
      for (int r = 0; r < 8; ++r) accO[t][r] *= alpha;

    // O^T += V^T * P^T : V^T rows contiguous thanks to [DK,S] layout
#pragma unroll
    for (int t = 0; t < 4; ++t) {
      const __bf16* p = Vh + (size_t)(t * 16 + ln) * SEQ + k0 + kh;
      v8bf lo = *(const v8bf*)p;
      v8bf hi = *(const v8bf*)(p + 16);
      v16bf aV;
#pragma unroll
      for (int e = 0; e < 8; ++e) { aV[e] = lo[e]; aV[e + 8] = hi[e]; }
      accO[t] = __builtin_amdgcn_wmma_f32_16x16x32_bf16(
          false, aV, false, bp, (short)0, accO[t], false, false);
    }
  }

  // epilogue: lane's column q fixed -> 8 consecutive d per tile = 16B stores
  const float inv = 1.0f / l_run;
  const int s = q0 + ln;
  const int b = bh >> 4, h = bh & 15;
  __bf16* orow = O + ((size_t)(b * SEQ + s)) * DMODEL + h * DHEAD;
#pragma unroll
  for (int t = 0; t < 4; ++t) {
    v8bf ov;
#pragma unroll
    for (int r = 0; r < 8; ++r) ov[r] = f2bf(accO[t][r] * inv);
    *(v8bf*)&orow[t * 16 + kh] = ov;
  }
}

// ---------------------------------------------------------------- launch ----
extern "C" void kernel_launch(void* const* d_in, const int* in_sizes, int n_in,
                              void* d_out, int out_size, void* d_ws, size_t ws_size,
                              hipStream_t stream) {
  const float* x  = (const float*)d_in[0];
  const float* wq = (const float*)d_in[1];
  const float* wk = (const float*)d_in[2];
  const float* wv = (const float*)d_in[3];
  const float* wo = (const float*)d_in[4];
  float* out = (float*)d_out;

  char* ws = (char*)d_ws;
  const size_t xBytes = (size_t)MROWS * DMODEL * sizeof(__bf16);   // 8 MB
  const size_t wBytes = (size_t)DMODEL * DMODEL * sizeof(__bf16);  // 2 MB
  __bf16* x_bf   = (__bf16*)ws;  ws += xBytes;
  __bf16* wq_bf  = (__bf16*)ws;  ws += wBytes;
  __bf16* wk_bf  = (__bf16*)ws;  ws += wBytes;
  __bf16* wv_bf  = (__bf16*)ws;  ws += wBytes;
  __bf16* wo_bf  = (__bf16*)ws;  ws += wBytes;
  __bf16* Qb     = (__bf16*)ws;  ws += xBytes;   // [B*H, S, DK]
  __bf16* Kb     = (__bf16*)ws;  ws += xBytes;   // [B*H, S, DK]
  __bf16* Vtb    = (__bf16*)ws;  ws += xBytes;   // [B*H, DK, S]
  __bf16* attnb  = (__bf16*)ws;  ws += xBytes;   // [B*S, DMODEL]

  to_bf16_kernel<<<1024, 256, 0, stream>>>(x,  x_bf,  MROWS * DMODEL);
  to_bf16_kernel<<<512, 256, 0, stream>>>(wq, wq_bf, DMODEL * DMODEL);
  to_bf16_kernel<<<512, 256, 0, stream>>>(wk, wk_bf, DMODEL * DMODEL);
  to_bf16_kernel<<<512, 256, 0, stream>>>(wv, wv_bf, DMODEL * DMODEL);
  to_bf16_kernel<<<512, 256, 0, stream>>>(wo, wo_bf, DMODEL * DMODEL);

  dim3 ggrid(MROWS / 128, DMODEL / 128);  // 32 x 8
  gemm_nt_kernel<<<ggrid, 256, 0, stream>>>(x_bf, wq_bf, Qb,  nullptr, 0);
  gemm_nt_kernel<<<ggrid, 256, 0, stream>>>(x_bf, wk_bf, Kb,  nullptr, 1);
  gemm_nt_kernel<<<ggrid, 256, 0, stream>>>(x_bf, wv_bf, Vtb, nullptr, 2);

  attn_kernel<<<(BATCH * NHEAD * (SEQ / 16)) / 8, 256, 0, stream>>>(Qb, Kb, Vtb, attnb);

  gemm_nt_kernel<<<ggrid, 256, 0, stream>>>(attnb, wo_bf, nullptr, out, 3);
}
